// UTSTopPool_64690797412426
// MI455X (gfx1250) — matrix operand order repaired and checked by reading, not compiled, not run on hardware
//
#include <hip/hip_runtime.h>

// ---------------------------------------------------------------------------
// UTSTopPool for MI455X (gfx1250, wave32, WMMA)
//   G=32 graphs x NG=4096 nodes, D=32, UTS_DIM=128, HID=64, k=2048, E=2^21
// ---------------------------------------------------------------------------

typedef __attribute__((ext_vector_type(16))) _Float16 v16h;
typedef __attribute__((ext_vector_type(8)))  _Float16 v8h;
typedef __attribute__((ext_vector_type(8)))  float    v8f;

#define DFEAT 32
#define UTSD  128
#define HIDD  64

// -------- float atomic max/min via sign-aware integer atomics --------------
__device__ __forceinline__ void atomicMaxF(float* addr, float v) {
    if (v >= 0.0f) atomicMax((int*)addr, __float_as_int(v));
    else           atomicMin((unsigned int*)addr, __float_as_uint(v));
}
__device__ __forceinline__ void atomicMinF(float* addr, float v) {
    if (v >= 0.0f) atomicMin((int*)addr, __float_as_int(v));
    else           atomicMax((unsigned int*)addr, __float_as_uint(v));
}

// -------- kernel 0: init stats with self-contribution, deg=0, remap=-1 -----
__global__ void uts_init_kernel(const float* __restrict__ x,
                                float* __restrict__ deg,
                                float* __restrict__ nsum, float* __restrict__ nsq,
                                float* __restrict__ mxp,  float* __restrict__ mnp,
                                int* __restrict__ remap, int N) {
    int i = blockIdx.x * blockDim.x + threadIdx.x;
    if (i >= N * DFEAT) return;
    float v = x[i];
    nsum[i] = v; nsq[i] = v * v; mxp[i] = v; mnp[i] = v;
    if ((i & (DFEAT - 1)) == 0) {
        int node = i / DFEAT;
        deg[node] = 0.0f;
        remap[node] = -1;
    }
}

// -------- kernel 1: edge scatter, one wave per edge, one lane per feature --
__global__ __launch_bounds__(256)
void uts_scatter_kernel(const int* __restrict__ ei, const float* __restrict__ x,
                        float* __restrict__ deg,
                        float* __restrict__ nsum, float* __restrict__ nsq,
                        float* __restrict__ mxp,  float* __restrict__ mnp, int E) {
    int gtid = blockIdx.x * blockDim.x + threadIdx.x;
    int lane = threadIdx.x & 31;
    int wid  = gtid >> 5;
    int nw   = (gridDim.x * blockDim.x) >> 5;
    for (int e = wid; e < E; e += nw) {
        int s = ei[e];
        int d = ei[E + e];
        float xs = x[s * DFEAT + lane];     // 128B coalesced per wave
        float xd = x[d * DFEAT + lane];
        atomicAdd(&nsum[d * DFEAT + lane], xs);
        atomicAdd(&nsum[s * DFEAT + lane], xd);
        atomicAdd(&nsq [d * DFEAT + lane], xs * xs);
        atomicAdd(&nsq [s * DFEAT + lane], xd * xd);
        atomicMaxF(&mxp[d * DFEAT + lane], xs);
        atomicMaxF(&mxp[s * DFEAT + lane], xd);
        atomicMinF(&mnp[d * DFEAT + lane], xs);
        atomicMinF(&mnp[s * DFEAT + lane], xd);
        if (lane == 0) {
            atomicAdd(&deg[s], 1.0f);
            atomicAdd(&deg[d], 1.0f);
        }
    }
}

// -------- kernel 2: uts stats -> (N,128)x(128,64) WMMA -> relu -> x w2 -> sigmoid
// Block = 256 threads = 8 waves; each wave scores 16 nodes with 16 WMMAs.
__global__ __launch_bounds__(256)
void uts_score_kernel(const float* __restrict__ deg,
                      const float* __restrict__ nsum, const float* __restrict__ nsq,
                      const float* __restrict__ mxp,  const float* __restrict__ mnp,
                      const float* __restrict__ w1, const float* __restrict__ b1,
                      const float* __restrict__ w2, const float* __restrict__ b2,
                      float* __restrict__ scores) {
    __shared__ alignas(64) _Float16 wtr[HIDD * UTSD];      // w1 transposed [col][k], 16KB
    __shared__ alignas(64) _Float16 utsT[8][16 * UTSD];    // per-wave 16x128 tile, 32KB
    __shared__ float b1s[HIDD];
    __shared__ float w2s[HIDD];

    int tid = threadIdx.x;
    for (int e = tid; e < UTSD * HIDD; e += 256) {
        int k = e / HIDD, c = e % HIDD;                    // w1 is [k][c]
        wtr[c * UTSD + k] = (_Float16)w1[e];
    }
    if (tid < HIDD) { b1s[tid] = b1[tid]; w2s[tid] = w2[tid]; }

    int wave = tid >> 5, lane = tid & 31;
    int row = lane & 15, hi = lane >> 4;
    int nodeBase = blockIdx.x * 128 + wave * 16;

    // --- build uts tile in f16: lane == feature (D==32) ---
    _Float16* U = utsT[wave];
    for (int n = 0; n < 16; ++n) {
        int node = nodeBase + n;
        float hood = deg[node] + 1.0f;
        int off = node * DFEAT + lane;
        float s = nsum[off], q = nsq[off];
        float mean = s / hood;
        float var  = fmaxf(q / hood - mean * mean, 0.0f);
        float sd   = sqrtf(var);
        float vmx = mxp[off], vmn = mnp[off];
        if (hood < 3.0f) { mean = 0.0f; sd = 0.0f; vmx = 0.0f; vmn = 0.0f; }
        U[n * UTSD +       lane] = (_Float16)mean;
        U[n * UTSD +  32 + lane] = (_Float16)sd;
        U[n * UTSD +  64 + lane] = (_Float16)vmx;
        U[n * UTSD +  96 + lane] = (_Float16)vmn;
    }
    __syncthreads();

    // --- 16x64 = sum_kc A(16x32) x B(32x16) tiles via v_wmma_f32_16x16x32_f16 ---
    v8f accs[4] = {v8f{}, v8f{}, v8f{}, v8f{}};
    #pragma unroll
    for (int kc = 0; kc < 4; ++kc) {
        // A fragment: row = lane%16; contiguous K = 16*hi + s (matches B)
        v16h a = *(const v16h*)&U[row * UTSD + kc * 32 + 16 * hi];
        #pragma unroll
        for (int nt = 0; nt < 4; ++nt) {
            int col = nt * 16 + row;
            v16h b = *(const v16h*)&wtr[col * UTSD + kc * 32 + 16 * hi];
            accs[nt] = __builtin_amdgcn_wmma_f32_16x16x32_f16(
                false, a, false, b, (short)0, accs[nt], false, false);
        }
    }

    // --- epilogue: h = relu(acc + b1); partial dot with w2; 16-lane butterfly ---
    float partial[8] = {0, 0, 0, 0, 0, 0, 0, 0};
    #pragma unroll
    for (int nt = 0; nt < 4; ++nt) {
        int col = nt * 16 + row;
        float bb = b1s[col], ww = w2s[col];
        #pragma unroll
        for (int r = 0; r < 8; ++r) {
            float h = fmaxf(accs[nt][r] + bb, 0.0f);
            partial[r] += h * ww;
        }
    }
    #pragma unroll
    for (int m = 1; m < 16; m <<= 1) {
        #pragma unroll
        for (int r = 0; r < 8; ++r)
            partial[r] += __shfl_xor(partial[r], m, 16);
    }
    if (row == 0) {
        float bias2 = b2[0];
        #pragma unroll
        for (int r = 0; r < 8; ++r) {
            int node = nodeBase + hi * 8 + r;   // C layout: VGPR r -> row r + 8*hi
            float sc = partial[r] + bias2;
            scores[node] = 1.0f / (1.0f + __expf(-sc));
        }
    }
}

// -------- kernel 3: per-graph descending bitonic sort of 4096 scores -------
__global__ __launch_bounds__(1024)
void uts_topk_kernel(const float* __restrict__ scores,
                     int* __restrict__ permI, float* __restrict__ permF,
                     float* __restrict__ batchF, int NG, int K) {
    __shared__ float sv[4096];
    __shared__ int   si[4096];
    int g = blockIdx.x, tid = threadIdx.x;
    for (int i = tid; i < NG; i += blockDim.x) { sv[i] = scores[g * NG + i]; si[i] = i; }
    __syncthreads();
    for (int size = 2; size <= NG; size <<= 1) {
        for (int stride = size >> 1; stride > 0; stride >>= 1) {
            for (int t = tid; t < (NG >> 1); t += blockDim.x) {
                int i = (t << 1) - (t & (stride - 1));
                int j = i + stride;
                bool asc = (i & size) != 0;          // overall descending
                float vi = sv[i], vj = sv[j];
                bool sw = asc ? (vi > vj) : (vi < vj);
                if (sw) {
                    sv[i] = vj; sv[j] = vi;
                    int ti = si[i]; si[i] = si[j]; si[j] = ti;
                }
            }
            __syncthreads();
        }
    }
    for (int t = tid; t < K; t += blockDim.x) {
        int node = g * NG + si[t];
        int oi = g * K + t;
        permI[oi]  = node;
        permF[oi]  = (float)node;
        batchF[oi] = (float)g;
    }
}

// -------- kernel 4: scatter remap[perm[i]] = i --------
__global__ void uts_remap_kernel(const int* __restrict__ permI,
                                 int* __restrict__ remap, int M) {
    int i = blockIdx.x * blockDim.x + threadIdx.x;
    if (i < M) remap[permI[i]] = i;
}

// -------- kernel 5: x_pool = relu(x[perm] @ wp + bp) via WMMA --------------
__global__ __launch_bounds__(256)
void uts_pool_kernel(const float* __restrict__ x, const int* __restrict__ permI,
                     const float* __restrict__ wp, const float* __restrict__ bp,
                     float* __restrict__ xp) {
    __shared__ alignas(64) _Float16 wpt[DFEAT * DFEAT];    // wp transposed [col][k]
    __shared__ alignas(64) _Float16 xt[8][16 * DFEAT];     // per-wave 16x32 tile
    __shared__ float bps[DFEAT];

    int tid = threadIdx.x;
    for (int e = tid; e < DFEAT * DFEAT; e += 256) {
        int k = e / DFEAT, c = e % DFEAT;                  // wp is [k][c]
        wpt[c * DFEAT + k] = (_Float16)wp[e];
    }
    if (tid < DFEAT) bps[tid] = bp[tid];

    int wave = tid >> 5, lane = tid & 31;
    int row = lane & 15, hi = lane >> 4;
    int rowBase = blockIdx.x * 128 + wave * 16;

    _Float16* X = xt[wave];
    for (int n = 0; n < 16; ++n) {
        int node = permI[rowBase + n];
        X[n * DFEAT + lane] = (_Float16)x[node * DFEAT + lane];
    }
    __syncthreads();

    v16h a = *(const v16h*)&X[row * DFEAT + 16 * hi];

    #pragma unroll
    for (int nt = 0; nt < 2; ++nt) {
        int col = nt * 16 + row;
        v16h b = *(const v16h*)&wpt[col * DFEAT + 16 * hi];
        v8f acc = {};
        acc = __builtin_amdgcn_wmma_f32_16x16x32_f16(
            false, a, false, b, (short)0, acc, false, false);
        float bb = bps[col];
        #pragma unroll
        for (int r = 0; r < 8; ++r) {
            int orow = rowBase + 8 * hi + r;
            xp[orow * DFEAT + col] = fmaxf(acc[r] + bb, 0.0f);
        }
    }
}

// -------- kernel 6: edge remap + keep mask, outputs as float ---------------
__global__ void uts_edgeout_kernel(const int* __restrict__ ei,
                                   const int* __restrict__ remap,
                                   float* __restrict__ ep, int E) {
    int e = blockIdx.x * blockDim.x + threadIdx.x;
    if (e >= E) return;
    int rs = remap[ei[e]];
    int rd = remap[ei[E + e]];
    bool keep = (rs >= 0) && (rd >= 0);
    ep[e]     = (float)(keep ? rs : -1);
    ep[E + e] = (float)(keep ? rd : -1);
}

// ---------------------------------------------------------------------------
extern "C" void kernel_launch(void* const* d_in, const int* in_sizes, int n_in,
                              void* d_out, int out_size, void* d_ws, size_t ws_size,
                              hipStream_t stream) {
    const float* x   = (const float*)d_in[0];
    const int*   ei  = (const int*)  d_in[1];
    const float* w1  = (const float*)d_in[3];
    const float* b1  = (const float*)d_in[4];
    const float* w2  = (const float*)d_in[5];
    const float* b2  = (const float*)d_in[6];
    const float* wp  = (const float*)d_in[7];
    const float* bpv = (const float*)d_in[8];

    const int N  = in_sizes[2];          // 131072
    const int E  = in_sizes[1] / 2;      // 2097152
    const int G  = 32;
    const int NG = N / G;                // 4096
    const int K  = NG / 2;               // 2048
    const int M  = G * K;                // 65536 pooled nodes

    // workspace layout (floats): deg | nsum | nsq | mx | mn | scores | permI | remap
    float* ws     = (float*)d_ws;
    float* deg    = ws;
    float* nsum   = deg  + N;
    float* nsq    = nsum + (size_t)N * DFEAT;
    float* mxp    = nsq  + (size_t)N * DFEAT;
    float* mnp    = mxp  + (size_t)N * DFEAT;
    float* scores = mnp  + (size_t)N * DFEAT;
    int*   permI  = (int*)(scores + N);
    int*   remap  = permI + M;

    // output layout: x_pool | edge_index_pool | batch_pool | perm  (all float)
    float* xp     = (float*)d_out;
    float* ep     = xp + (size_t)M * DFEAT;
    float* bpool  = ep + (size_t)2 * E;
    float* pf     = bpool + M;

    uts_init_kernel<<<(N * DFEAT + 255) / 256, 256, 0, stream>>>(
        x, deg, nsum, nsq, mxp, mnp, remap, N);

    uts_scatter_kernel<<<2048, 256, 0, stream>>>(
        ei, x, deg, nsum, nsq, mxp, mnp, E);

    uts_score_kernel<<<N / 128, 256, 0, stream>>>(
        deg, nsum, nsq, mxp, mnp, w1, b1, w2, b2, scores);

    uts_topk_kernel<<<G, 1024, 0, stream>>>(scores, permI, pf, bpool, NG, K);

    uts_remap_kernel<<<(M + 255) / 256, 256, 0, stream>>>(permI, remap, M);

    uts_pool_kernel<<<M / 128, 256, 0, stream>>>(x, permI, wp, bpv, xp);

    uts_edgeout_kernel<<<(E + 255) / 256, 256, 0, stream>>>(ei, remap, ep, E);
}
